// CarafeGuidedUpsampler_73212012528112
// MI455X (gfx1250) — matrix operand order repaired and checked by reading, not compiled, not run on hardware
//
#include <hip/hip_runtime.h>
#include <math.h>

typedef __attribute__((ext_vector_type(16))) _Float16 v16h;
typedef __attribute__((ext_vector_type(8)))  float    v8f;

#define B_    4
#define DIMC  384
#define CIN   387
#define KP    416      // CIN padded to multiple of 32
#define COMPC 64
#define KE    576      // 64*9, multiple of 32
#define ME    36
#define MEP   64       // ME padded so M % 64 == 0
#define MT    4        // 16-row m-subtiles per block (block tile 64 x 128)

// ---------------- weight / bias packing ----------------

__global__ void pack_w_f16(const float* __restrict__ W, _Float16* __restrict__ A,
                           int Mreal, int Mpad, int K, int Kpad) {
    int i = blockIdx.x * 256 + threadIdx.x;
    int tot = Mpad * Kpad;
    if (i >= tot) return;
    int m = i / Kpad, k = i % Kpad;
    float v = (m < Mreal && k < K) ? W[(size_t)m * K + k] : 0.f;
    A[i] = (_Float16)v;
}

__global__ void pad_bias_f32(const float* __restrict__ b, float* __restrict__ o,
                             int Mreal, int Mpad) {
    int i = blockIdx.x * 256 + threadIdx.x;
    if (i >= Mpad) return;
    o[i] = (i < Mreal) ? b[i] : 0.f;
}

// ---------------- stage-input builders ----------------

// source [B][384][196] f32 -> rows 0..383 of X0 [B][KP][Np]
__global__ void pack_src(const float* __restrict__ src, _Float16* __restrict__ X,
                         int N, int Np) {
    int i = blockIdx.x * 256 + threadIdx.x;
    int tot = B_ * DIMC * N;
    if (i >= tot) return;
    int n = i % N;
    int c = (i / N) % DIMC;
    int b = i / (N * DIMC);
    X[((size_t)b * KP + c) * Np + n] = (_Float16)src[i];
}

// bilinear-resize guidance into rows 384..386 of X (h x h), zero rows 387..KP-1
__global__ void guide_rows(const float* __restrict__ g, _Float16* __restrict__ X,
                           int h, int Np) {
    const int w = h, N = h * w;
    const int rows = KP - DIMC; // 32
    int idx = blockIdx.x * 256 + threadIdx.x;
    int tot = B_ * rows * N;
    if (idx >= tot) return;
    int n = idx % N;
    int r = (idx / N) % rows;
    int b = idx / (N * rows);
    float val = 0.f;
    if (r < 3) {
        int i = n / w, j = n % w;
        float scale = 224.f / (float)h;
        float sy = ((float)i + 0.5f) * scale - 0.5f;
        float sx = ((float)j + 0.5f) * scale - 0.5f;
        int y0 = (int)floorf(sy), x0 = (int)floorf(sx);
        float fy = sy - (float)y0, fx = sx - (float)x0;
        int y0c = y0 < 0 ? 0 : (y0 > 223 ? 223 : y0);
        int y1 = y0 + 1; int y1c = y1 < 0 ? 0 : (y1 > 223 ? 223 : y1);
        int x0c = x0 < 0 ? 0 : (x0 > 223 ? 223 : x0);
        int x1 = x0 + 1; int x1c = x1 < 0 ? 0 : (x1 > 223 ? 223 : x1);
        const float* gb = g + ((size_t)b * 3 + r) * 224 * 224;
        float v00 = gb[y0c * 224 + x0c], v01 = gb[y0c * 224 + x1c];
        float v10 = gb[y1c * 224 + x0c], v11 = gb[y1c * 224 + x1c];
        val = (1.f - fy) * ((1.f - fx) * v00 + fx * v01)
            +        fy  * ((1.f - fx) * v10 + fx * v11);
    }
    X[((size_t)b * KP + DIMC + r) * Np + n] = (_Float16)val;
}

// zero the pad columns [N, Np) of all `rows` rows
__global__ void zero_pad_cols(_Float16* __restrict__ X, int rows, int N, int Np) {
    int pad = Np - N;
    int idx = blockIdx.x * 256 + threadIdx.x;
    int tot = B_ * rows * pad;
    if (idx >= tot) return;
    int p = idx % pad;
    int r = (idx / pad) % rows;
    int b = idx / (pad * rows);
    X[((size_t)b * rows + r) * Np + N + p] = (_Float16)0.f;
}

// ---------------- WMMA GEMM: Y[b][M][Np] = A[M][Kpad] * X[b][Kpad][Np] + bias ----------
// 256 threads = 8 waves. Block tile 64(M) x 128(N): 4 WMMA accumulators per wave.
// LDS tiles are kept in WMMA *fragment* layout so reads are 32B vector loads.
// Requires: M % 64 == 0, Kpad % 32 == 0, Np % 128 == 0 (all padded). No guards.

__global__ __launch_bounds__(256) void gemm_wmma(
    const _Float16* __restrict__ A, const _Float16* __restrict__ X,
    const float* __restrict__ bias, float* __restrict__ Y,
    int M, int Np, int Kpad)
{
    __shared__ v16h AsF[MT * 32];   // [m-subtile][lane] fragment, 4 KB
    __shared__ v16h BsF[8 * 32];    // [wave][lane] fragment, 8 KB
    _Float16* AsH = (_Float16*)AsF;
    _Float16* BsH = (_Float16*)BsF;

    const int tid  = threadIdx.x;
    const int wv   = tid >> 5;
    const int lane = tid & 31;
    const int half = lane >> 4;
    const int nl   = lane & 15;
    const int m0   = blockIdx.y * (16 * MT);
    const int n0   = blockIdx.x * 128;

    const _Float16* Xb = X + (size_t)blockIdx.z * Kpad * Np;
    float*          Yb = Y + (size_t)blockIdx.z * M * Np;

    v8f acc[MT] = {};

    for (int k0 = 0; k0 < Kpad; k0 += 32) {
        { // fill A fragments: thread t -> lane la, element pair ep (dword moves)
            int la = tid >> 3;            // 0..31
            int ep = (tid & 7) * 2;       // 0,2,..,14
            int ma = la & 15, ha = la >> 4;
            int v  = ep >> 1;
            int k  = ((v >> 2) << 4) + (ha << 3) + ((v & 3) << 1);
#pragma unroll
            for (int mt = 0; mt < MT; ++mt) {
                const _Float16* s = A + (size_t)(m0 + mt * 16 + ma) * Kpad + k0 + k;
                AsH[(mt * 32 + la) * 16 + ep]     = s[0];
                AsH[(mt * 32 + la) * 16 + ep + 1] = s[1];
            }
        }
        { // fill B fragments: thread t -> K-row r, n 16-group seg (== target wave)
            int r = tid >> 3, seg = tid & 7;
            const _Float16* s = Xb + (size_t)(k0 + r) * Np + n0 + seg * 16;
            if (k0 + 32 < Kpad)
                __builtin_prefetch((const void*)(s + (size_t)32 * Np), 0, 0);
            int laneBase = ((r >> 3) & 1) * 16; // lane group: h = (k / 8) % 2
            int e = (r & 7) | ((r >> 4) << 3);  // fragment element: k%8 + 8*(k/16)
#pragma unroll
            for (int u = 0; u < 16; ++u)
                BsH[(seg * 32 + laneBase + u) * 16 + e] = s[u];
        }
        __syncthreads();

        v16h bf = BsF[wv * 32 + lane];
#pragma unroll
        for (int mt = 0; mt < MT; ++mt) {
            v16h af = AsF[mt * 32 + lane];
            acc[mt] = __builtin_amdgcn_wmma_f32_16x16x32_f16(
                false, af, false, bf, (short)0, acc[mt], false, false);
        }
        __syncthreads();
    }

#pragma unroll
    for (int mt = 0; mt < MT; ++mt)
#pragma unroll
        for (int r = 0; r < 8; ++r) {    // D layout: VGPR r -> M = r + 8*half
            int gm = m0 + mt * 16 + half * 8 + r;
            Yb[(size_t)gm * Np + n0 + wv * 16 + nl] = acc[mt][r] + bias[gm];
        }
}

// ---------------- im2col for the 3x3 encoder conv ----------------
// comp [B][64][Np] f32 -> Xm [B][576][Np] f16, row r = c*9 + (di*3+dj); pad cols zeroed

__global__ void im2col9(const float* __restrict__ comp, _Float16* __restrict__ Xm,
                        int h, int Np) {
    const int w = h, N = h * w;
    int idx = blockIdx.x * 256 + threadIdx.x;
    int tot = B_ * KE * Np;
    if (idx >= tot) return;
    int n = idx % Np;
    int r = (idx / Np) % KE;
    int b = idx / (Np * KE);
    float v = 0.f;
    if (n < N) {
        int c = r / 9, t = r % 9;
        int di = t / 3 - 1, dj = t % 3 - 1;
        int i = n / w + di, j = n % w + dj;
        if (i >= 0 && i < h && j >= 0 && j < w)
            v = comp[((size_t)b * COMPC + c) * Np + i * w + j];
    }
    Xm[idx] = (_Float16)v;
}

// ---------------- CARAFE: pixel-shuffle + softmax + 9-tap reassembly ----------------
// one block per output pixel, thread = channel. mask [B][MEP][Np] f32,
// Xs [B][KP][Np] f16 -> Xn rows 0..383 [B][KP][Npo] f16

__global__ __launch_bounds__(384) void carafe_apply(
    const _Float16* __restrict__ Xs, const float* __restrict__ mask,
    _Float16* __restrict__ Xn, int h, int Np, int Npo)
{
    const int w = h;
    const int ow = 2 * w;
    int oj = blockIdx.x, oi = blockIdx.y, b = blockIdx.z;
    int i = oi >> 1, j = oj >> 1, p = oi & 1, q = oj & 1;

    __shared__ float sm[12];
    __shared__ float se[12];
    int t = threadIdx.x;
    if (t < 9) {
        int ch = t * 4 + p * 2 + q;                 // pixel-shuffle
        sm[t] = mask[((size_t)b * MEP + ch) * Np + i * w + j];
    }
    __syncthreads();
    if (t < 9) {
        float mx = sm[0];
#pragma unroll
        for (int u = 1; u < 9; ++u) mx = fmaxf(mx, sm[u]);
        se[t] = expf(sm[t] - mx);
    }
    __syncthreads();
    float ssum = 0.f;
#pragma unroll
    for (int u = 0; u < 9; ++u) ssum += se[u];
    float inv = 1.f / ssum;

    int c = t;                                       // 0..383
    const _Float16* xb = Xs + ((size_t)b * KP + c) * Np;
    float acc = 0.f;
#pragma unroll
    for (int di = -1; di <= 1; ++di)
#pragma unroll
        for (int dj = -1; dj <= 1; ++dj) {
            int ii = i + di, jj = j + dj;
            float xv = (ii >= 0 && ii < h && jj >= 0 && jj < w)
                     ? (float)xb[ii * w + jj] : 0.f;
            acc += se[(di + 1) * 3 + (dj + 1)] * xv;
        }
    Xn[((size_t)b * KP + c) * Npo + oi * ow + oj] = (_Float16)(acc * inv);
}

// ---------------- host launch ----------------

static inline int cdiv(int a, int b) { return (a + b - 1) / b; }

extern "C" void kernel_launch(void* const* d_in, const int* in_sizes, int n_in,
                              void* d_out, int out_size, void* d_ws, size_t ws_size,
                              hipStream_t stream) {
    (void)in_sizes; (void)n_in; (void)out_size; (void)ws_size;
    const float* source   = (const float*)d_in[0];
    const float* guidance = (const float*)d_in[1];
    const float* cw[4] = {(const float*)d_in[2],  (const float*)d_in[6],
                          (const float*)d_in[10], (const float*)d_in[14]};
    const float* cb[4] = {(const float*)d_in[3],  (const float*)d_in[7],
                          (const float*)d_in[11], (const float*)d_in[15]};
    const float* ew[4] = {(const float*)d_in[4],  (const float*)d_in[8],
                          (const float*)d_in[12], (const float*)d_in[16]};
    const float* eb[4] = {(const float*)d_in[5],  (const float*)d_in[9],
                          (const float*)d_in[13], (const float*)d_in[17]};
    const float* proj_w = (const float*)d_in[18];
    const float* proj_b = (const float*)d_in[19];
    float* out = (float*)d_out;

    static const int HS[5] = {14, 28, 56, 112, 224};
    static const int NP[5] = {256, 896, 3200, 12544, 50176}; // N padded to %128
    int NS[5]; for (int s = 0; s < 5; ++s) NS[s] = HS[s] * HS[s];

    char* ws = (char*)d_ws;
    size_t off = 0;
    auto take = [&](size_t bytes) -> size_t {
        size_t o = off; off += (bytes + 255) & ~(size_t)255; return o;
    };
    size_t oAwc[4], oAwe[4], oBe[4], oX[5];
    for (int s = 0; s < 4; ++s) oAwc[s] = take((size_t)COMPC * KP * 2);
    for (int s = 0; s < 4; ++s) oAwe[s] = take((size_t)MEP * KE * 2);
    size_t oAwp = take((size_t)DIMC * KP * 2);
    for (int s = 0; s < 4; ++s) oBe[s] = take((size_t)MEP * 4);
    for (int s = 0; s < 5; ++s) oX[s] = take((size_t)B_ * KP * NP[s] * 2);
    size_t oComp = take((size_t)B_ * COMPC * NP[3] * 4);
    size_t oXm   = take((size_t)B_ * KE * NP[3] * 2);
    size_t oMk   = take((size_t)B_ * MEP * NP[3] * 4);

    // --- pack weights/biases to f16 / padded ---
    for (int s = 0; s < 4; ++s) {
        int tot = COMPC * KP;
        pack_w_f16<<<cdiv(tot, 256), 256, 0, stream>>>(
            cw[s], (_Float16*)(ws + oAwc[s]), COMPC, COMPC, CIN, KP);
        tot = MEP * KE;
        pack_w_f16<<<cdiv(tot, 256), 256, 0, stream>>>(
            ew[s], (_Float16*)(ws + oAwe[s]), ME, MEP, KE, KE);
        pad_bias_f32<<<1, 256, 0, stream>>>(eb[s], (float*)(ws + oBe[s]), ME, MEP);
    }
    {
        int tot = DIMC * KP;
        pack_w_f16<<<cdiv(tot, 256), 256, 0, stream>>>(
            proj_w, (_Float16*)(ws + oAwp), DIMC, DIMC, CIN, KP);
    }

    // --- stage 0 input: source + resized guidance + pads ---
    {
        int tot = B_ * DIMC * NS[0];
        pack_src<<<cdiv(tot, 256), 256, 0, stream>>>(
            source, (_Float16*)(ws + oX[0]), NS[0], NP[0]);
        tot = B_ * (KP - DIMC) * NS[0];
        guide_rows<<<cdiv(tot, 256), 256, 0, stream>>>(
            guidance, (_Float16*)(ws + oX[0]), HS[0], NP[0]);
        tot = B_ * KP * (NP[0] - NS[0]);
        zero_pad_cols<<<cdiv(tot, 256), 256, 0, stream>>>(
            (_Float16*)(ws + oX[0]), KP, NS[0], NP[0]);
    }

    // --- four CARAFE stages ---
    for (int s = 0; s < 4; ++s) {
        int h = HS[s], N = NS[s], Np = NP[s];
        int oh = HS[s + 1], Npo = NP[s + 1];
        // compress: [64 x 416] * [416 x Np]
        gemm_wmma<<<dim3(Np / 128, COMPC / 64, B_), 256, 0, stream>>>(
            (const _Float16*)(ws + oAwc[s]), (const _Float16*)(ws + oX[s]),
            cb[s], (float*)(ws + oComp), COMPC, Np, KP);
        // im2col 3x3 (zero-fills pad columns too)
        int tot = B_ * KE * Np;
        im2col9<<<cdiv(tot, 256), 256, 0, stream>>>(
            (const float*)(ws + oComp), (_Float16*)(ws + oXm), h, Np);
        // encoder: [64 x 576] * [576 x Np] (rows 36..63 are zero padding)
        gemm_wmma<<<dim3(Np / 128, MEP / 64, B_), 256, 0, stream>>>(
            (const _Float16*)(ws + oAwe[s]), (const _Float16*)(ws + oXm),
            (const float*)(ws + oBe[s]), (float*)(ws + oMk), MEP, Np, KE);
        // CARAFE reassembly -> next stage rows 0..383
        carafe_apply<<<dim3(oh, oh, B_), 384, 0, stream>>>(
            (const _Float16*)(ws + oX[s]), (const float*)(ws + oMk),
            (_Float16*)(ws + oX[s + 1]), h, Np, Npo);
        // next stage rows 384..415 (guidance + zero rows) and pad columns
        tot = B_ * (KP - DIMC) * NS[s + 1];
        guide_rows<<<cdiv(tot, 256), 256, 0, stream>>>(
            guidance, (_Float16*)(ws + oX[s + 1]), oh, Npo);
        if (Npo > NS[s + 1]) {
            tot = B_ * KP * (Npo - NS[s + 1]);
            zero_pad_cols<<<cdiv(tot, 256), 256, 0, stream>>>(
                (_Float16*)(ws + oX[s + 1]), KP, NS[s + 1], Npo);
        }
    }

    // --- final projection straight into d_out (NCHW fp32, N = 50176 = 392*128) ---
    gemm_wmma<<<dim3(NP[4] / 128, DIMC / 64, B_), 256, 0, stream>>>(
        (const _Float16*)(ws + oAwp), (const _Float16*)(ws + oX[4]),
        proj_b, out, DIMC, NP[4], KP);
}
